// NEG_loss_44461501448871
// MI455X (gfx1250) — matrix operand order repaired and checked by reading, not compiled, not run on hardware
//
#include <hip/hip_runtime.h>
#include <hip/hip_bf16.h>
#include <math.h>

typedef __attribute__((ext_vector_type(16))) _Float16 v16h;
typedef __attribute__((ext_vector_type(8)))  float    v8f;

#define VV 50000
#define EE 256
#define BB 4096
#define WW 5
#define SS 15
#define NROWS (BB * WW)   // 20480
#define WAVES_PER_BLOCK 8 // 256 threads / wave32

// One wave32 per row n. 16 dot-products of length 256 via 8x V_WMMA_F32_16X16X32_F16.
__global__ __launch_bounds__(256)
void neg_loss_wmma_kernel(const float* __restrict__ in_embed,
                          const float* __restrict__ out_embed,
                          const int*   __restrict__ input_labels,
                          const int*   __restrict__ out_labels,
                          const int*   __restrict__ noise_labels,
                          float*       __restrict__ partials) {
    const int wave_in_blk = threadIdx.x >> 5;
    const int lane        = threadIdx.x & 31;
    const int n           = blockIdx.x * WAVES_PER_BLOCK + wave_in_blk; // row id, < NROWS by grid sizing

    const int jcol = lane & 15;  // which of the 16 target vectors this lane owns (B-matrix column)
    const int half = lane >> 4;  // 0: low K half, 1: high K half

    // in_idx = input_labels[n % B] (jnp.tile semantics)
    const int in_idx = input_labels[n & (BB - 1)];
    // target row index: column 0 -> positive (out_labels), columns 1..15 -> noise rows
    const int nz     = (jcol == 0) ? 0 : (jcol - 1);
    const int t_noise = noise_labels[n * SS + nz];
    const int t_idx   = (jcol == 0) ? out_labels[n] : t_noise;

    const float* __restrict__ irow = in_embed  + (size_t)in_idx * EE;
    const float* __restrict__ trow = out_embed + (size_t)t_idx  * EE;

    v8f acc = {};
    #pragma unroll
    for (int kc = 0; kc < EE / 32; ++kc) {
        const int kbase = kc * 32;

        // A matrix 16x32 f16: inp chunk replicated across all 16 M-rows.
        // lane<16: halves 0..7 = K 0..7, halves 8..15 = K 16..23
        // lane>=16: halves 0..7 = K 8..15, halves 8..15 = K 24..31
        v16h a;
        const int abase = kbase + half * 8;
        #pragma unroll
        for (int t = 0; t < 8; ++t) {
            a[t]     = (_Float16)irow[abase + t];
            a[8 + t] = (_Float16)irow[abase + 16 + t];
        }

        // B matrix 32x16 f16: column N = lane%16 is target vector jcol.
        // lane<16 holds K 0..7,16..23; lane>=16 holds K 8..15,24..31 (same order as A)
        v16h b;
        const int bbase = kbase + half * 8;
        #pragma unroll
        for (int t = 0; t < 8; ++t) {
            b[t]     = (_Float16)trow[bbase + t];
            b[8 + t] = (_Float16)trow[bbase + 16 + t];
        }

        acc = __builtin_amdgcn_wmma_f32_16x16x32_f16(
            /*neg_a=*/false, a, /*neg_b=*/false, b,
            /*c_mod=*/(short)0, acc, /*reuse_a=*/false, /*reuse_b=*/false);
    }

    // All 16 D rows are identical (A rows replicated). Per C/D layout, acc[0] on
    // this lane is D[row][lane%16] => d_j for j = jcol.
    float d = acc[0];
    // positive pair: log_sigmoid(+d); noise: log_sigmoid(-d)
    float x  = (jcol == 0) ? d : -d;
    // numerically stable log_sigmoid
    float ls = fminf(x, 0.0f) - log1pf(__expf(-fabsf(x)));

    // The two half-waves hold duplicate values; count only lanes 0..15.
    float contrib = (half == 0) ? ls : 0.0f;
    #pragma unroll
    for (int off = 16; off >= 1; off >>= 1)
        contrib += __shfl_xor(contrib, off, 32);

    if (lane == 0)
        partials[n] = contrib;
}

// Deterministic fixed-order reduction of the 20480 per-row partials.
__global__ __launch_bounds__(256)
void neg_loss_reduce_kernel(const float* __restrict__ partials,
                            float* __restrict__ out) {
    __shared__ float smem[256];
    float s = 0.0f;
    for (int i = threadIdx.x; i < NROWS; i += 256)
        s += partials[i];
    smem[threadIdx.x] = s;
    __syncthreads();
    #pragma unroll
    for (int st = 128; st >= 1; st >>= 1) {
        if ((int)threadIdx.x < st)
            smem[threadIdx.x] += smem[threadIdx.x + st];
        __syncthreads();
    }
    if (threadIdx.x == 0)
        out[0] = -smem[0] * (1.0f / (float)BB);
}

extern "C" void kernel_launch(void* const* d_in, const int* in_sizes, int n_in,
                              void* d_out, int out_size, void* d_ws, size_t ws_size,
                              hipStream_t stream) {
    const float* in_embed     = (const float*)d_in[0];
    const float* out_embed    = (const float*)d_in[1];
    const int*   input_labels = (const int*)d_in[2];
    const int*   out_labels   = (const int*)d_in[3];
    const int*   noise_labels = (const int*)d_in[4];
    // d_in[5] = num_sampled (compile-time constant SS here)

    float* partials = (float*)d_ws;       // NROWS floats = 80 KB
    float* out      = (float*)d_out;

    const int grid = NROWS / WAVES_PER_BLOCK; // 2560 blocks x 256 threads = 20480 waves
    neg_loss_wmma_kernel<<<grid, 256, 0, stream>>>(
        in_embed, out_embed, input_labels, out_labels, noise_labels, partials);
    neg_loss_reduce_kernel<<<1, 256, 0, stream>>>(partials, out);
}
